// SFCLayer_979252544080
// MI455X (gfx1250) — matrix operand order repaired and checked by reading, not compiled, not run on hardware
//
#include <hip/hip_runtime.h>

// ---------------------------------------------------------------------------
// SFC layer: drive = x @ W^T + b  (M=65536, N=512, K=512) + leaky-integrator
// scan over T=2048.
// K0: one-time W f32 -> bf16 hi/lo split into workspace.
// K1: WMMA bf16x3 GEMM; B-tiles double-buffered via async global->LDS
//     (ASYNCcnt); A-tiles converted f32->bf16 hi/lo in-flight; padded LDS
//     strides for conflict-free fragment reads; fused local scan over
//     128-row t-chunks; chunk summaries to workspace.
// K2: carry fixup: out[b, j*128+tl, n] += alpha^{tl+1} * carry_j.
// ---------------------------------------------------------------------------

typedef __attribute__((ext_vector_type(16))) __bf16 v16bf;
typedef __attribute__((ext_vector_type(8)))  float  v8f;
typedef unsigned short ushort_t;

union Frag { unsigned u[8]; v16bf v; };
union AccU { float f[8]; v8f v; };

__device__ __forceinline__ ushort_t f2bf_rne(float f) {
    unsigned u = __builtin_bit_cast(unsigned, f);
    unsigned r = u + 0x7FFFu + ((u >> 16) & 1u);
    return (ushort_t)(r >> 16);
}
__device__ __forceinline__ float bf2f(ushort_t h) {
    unsigned u = ((unsigned)h) << 16;
    return __builtin_bit_cast(float, u);
}

// ---- CDNA5 async global->LDS copy (ASYNCcnt path), with safe fallback -----
#if __has_builtin(__builtin_amdgcn_global_load_async_to_lds_b128)
#define HAVE_ASYNC_LDS 1
#else
#define HAVE_ASYNC_LDS 0
#endif

typedef __attribute__((__vector_size__(4 * sizeof(int)))) int v4i_t;
typedef __attribute__((address_space(1))) v4i_t* g_v4i_ptr;
typedef __attribute__((address_space(3))) v4i_t* l_v4i_ptr;

__device__ __forceinline__ void async_copy16(const void* g, void* l) {
#if HAVE_ASYNC_LDS
    __builtin_amdgcn_global_load_async_to_lds_b128(
        (g_v4i_ptr)(g), (l_v4i_ptr)(l), 0, 0);
#else
    *(uint4*)l = *(const uint4*)g;
#endif
}
__device__ __forceinline__ void async_wait0() {
#if HAVE_ASYNC_LDS
#if __has_builtin(__builtin_amdgcn_s_wait_asynccnt)
    __builtin_amdgcn_s_wait_asynccnt(0);
#else
    asm volatile("s_wait_asynccnt 0" ::: "memory");
#endif
#endif
}

#define BM 128
#define BN 128
#define BK 32
#define ASTR 38   // A row stride (ushorts): 76B -> bank stride 19, conflict-free
#define BSTR 40   // B row stride (ushorts): 80B -> 16B-aligned for async b128
#define TT 2048   // timesteps
#define NN 512    // OUT
#define KK 512    // IN
#define NCHUNK 16 // TT / BM

union SMem {
    struct {
        ushort_t Ah[BM][ASTR];     // 9728 B   (single-buffered, VALU-converted)
        ushort_t Al[BM][ASTR];     // 9728 B
        ushort_t Bh[2][BN][BSTR];  // 20480 B  (double-buffered, async-filled)
        ushort_t Bl[2][BN][BSTR];  // 20480 B
    } s;                           // 60416 B
    float C[BM][BN];               // 65536 B (epilogue scan tile)
};

// ---- K0: split W into bf16 hi/lo once --------------------------------------
__global__ __launch_bounds__(256)
void sfc_convert_w_k0(const float* __restrict__ W,
                      ushort_t* __restrict__ Wh, ushort_t* __restrict__ Wl) {
    int i = blockIdx.x * 256 + threadIdx.x;   // < 512*512
    float f = W[i];
    ushort_t h = f2bf_rne(f);
    Wh[i] = h;
    Wl[i] = f2bf_rne(f - bf2f(h));
}

// ---- K1: GEMM (bf16x3) + fused local scan ----------------------------------
__global__ __launch_bounds__(256)
void sfc_gemm_scan_k1(const float* __restrict__ x,
                      const ushort_t* __restrict__ Wh, const ushort_t* __restrict__ Wl,
                      const float* __restrict__ bias, const float* __restrict__ alpha,
                      float* __restrict__ out, float* __restrict__ S) {
    __shared__ SMem smem;

    const int tid  = threadIdx.x;
    const int bx   = blockIdx.x;           // n-tile: 0..3
    const int by   = blockIdx.y;           // m-tile: 0..511 (== b*16 + chunk)
    const int m0   = by * BM;
    const int n0   = bx * BN;

    const int lane = tid & 31;
    const int wid  = tid >> 5;
    const int wm   = wid >> 2;             // 0..1  (64 rows each)
    const int wn   = wid & 3;              // 0..3  (32 cols each)
    const int lm   = lane & 15;
    const int hi   = (lane >> 4) & 1;

    AccU acc[4][2];
    #pragma unroll
    for (int mi = 0; mi < 4; ++mi)
        #pragma unroll
        for (int ni = 0; ni < 2; ++ni)
            #pragma unroll
            for (int r = 0; r < 8; ++r) acc[mi][ni].f[r] = 0.0f;

    float4 vx[4];

    auto issue_B = [&](int buf, int ko) {
        #pragma unroll
        for (int h2 = 0; h2 < 2; ++h2) {
            int q  = tid + 256 * h2;       // 0..511
            int r  = q >> 2;               // row 0..127
            int sg = (q & 3) * 8;          // 8-ushort (16B) segment
            size_t go = (size_t)(n0 + r) * KK + ko + sg;
            async_copy16(Wh + go, &smem.s.Bh[buf][r][sg]);
            async_copy16(Wl + go, &smem.s.Bl[buf][r][sg]);
        }
    };
    auto load_x = [&](int ko) {
        #pragma unroll
        for (int i = 0; i < 4; ++i) {
            int q = tid + 256 * i;         // 0..1023
            int r = q >> 3;
            int c = (q & 7) * 4;
            vx[i] = *(const float4*)(x + (size_t)(m0 + r) * KK + ko + c);
        }
    };
    auto store_A = [&]() {
        #pragma unroll
        for (int i = 0; i < 4; ++i) {
            int q = tid + 256 * i;
            int r = q >> 3;
            int c = (q & 7) * 4;
            #pragma unroll
            for (int u = 0; u < 4; ++u) {
                float f = (&vx[i].x)[u];
                ushort_t h = f2bf_rne(f);
                smem.s.Ah[r][c + u] = h;
                smem.s.Al[r][c + u] = f2bf_rne(f - bf2f(h));
            }
        }
    };

    // --- prologue: stage tile 0 ---
    issue_B(0, 0);
    load_x(0);
    store_A();
    async_wait0();
    __syncthreads();

    // --- main loop: B double-buffered (async), A single-buffered ---
    for (int it = 0; it < KK / BK; ++it) {
        const int cur = it & 1;
        const int nxt = cur ^ 1;
        if (it < KK / BK - 1) {
            issue_B(nxt, (it + 1) * BK);   // async -> other B buffer
            load_x((it + 1) * BK);         // global f32 -> regs (overlaps WMMA)
        }

        // B fragments (contiguous k = 16*hi + s, same order as A)
        Frag bh[2], bl[2];
        #pragma unroll
        for (int ni = 0; ni < 2; ++ni) {
            int n  = wn * 32 + ni * 16 + lm;
            int kb = hi * 16;
            #pragma unroll
            for (int jj = 0; jj < 8; ++jj) {
                bh[ni].u[jj] = *(const unsigned*)&smem.s.Bh[cur][n][kb + 2 * jj];
                bl[ni].u[jj] = *(const unsigned*)&smem.s.Bl[cur][n][kb + 2 * jj];
            }
        }
        // A fragments + 3x WMMA (bf16 split-precision)
        #pragma unroll
        for (int mi = 0; mi < 4; ++mi) {
            Frag ah, al;
            int m = wm * 64 + mi * 16 + lm;
            #pragma unroll
            for (int jj = 0; jj < 8; ++jj) {
                int k = hi * 16 + 2 * jj;
                ah.u[jj] = *(const unsigned*)&smem.s.Ah[m][k];
                al.u[jj] = *(const unsigned*)&smem.s.Al[m][k];
            }
            #pragma unroll
            for (int ni = 0; ni < 2; ++ni) {
                acc[mi][ni].v = __builtin_amdgcn_wmma_f32_16x16x32_bf16(
                    false, ah.v, false, bh[ni].v, (short)0, acc[mi][ni].v, false, false);
                acc[mi][ni].v = __builtin_amdgcn_wmma_f32_16x16x32_bf16(
                    false, ah.v, false, bl[ni].v, (short)0, acc[mi][ni].v, false, false);
                acc[mi][ni].v = __builtin_amdgcn_wmma_f32_16x16x32_bf16(
                    false, al.v, false, bh[ni].v, (short)0, acc[mi][ni].v, false, false);
            }
        }

        __syncthreads();                    // all waves done reading A buffer
        if (it < KK / BK - 1) store_A();    // overwrite single A buffer
        async_wait0();                      // own async B writes complete
        __syncthreads();                    // all writes visible to all waves
    }

    // ---- epilogue: park tile in LDS ------------------------------------
    #pragma unroll
    for (int mi = 0; mi < 4; ++mi)
        #pragma unroll
        for (int ni = 0; ni < 2; ++ni)
            #pragma unroll
            for (int r = 0; r < 8; ++r) {
                int row = wm * 64 + mi * 16 + r + hi * 8;
                int col = wn * 32 + ni * 16 + lm;
                smem.C[row][col] = acc[mi][ni].f[r];
            }
    __syncthreads();

    // ---- bias + local leaky-integrator scan over the 128-row t-chunk ----
    if (tid < BN) {
        float a  = alpha[n0 + tid];
        float bb = bias[n0 + tid];
        float p  = 0.0f;
        for (int t = 0; t < BM; ++t) {
            p = fmaf(a, p, smem.C[t][tid] + bb);
            smem.C[t][tid] = p;
        }
    }
    __syncthreads();

    // ---- coalesced store of locally-scanned tile ------------------------
    #pragma unroll
    for (int i = 0; i < 16; ++i) {
        int q = tid + 256 * i;             // 0..4095
        int r = q >> 5;
        int c = (q & 31) * 4;
        *(float4*)(out + (size_t)(m0 + r) * NN + n0 + c) = *(const float4*)&smem.C[r][c];
    }
    // ---- chunk summary (scan value at last t of chunk) ------------------
    if (tid < BN) {
        S[(size_t)by * NN + n0 + tid] = smem.C[BM - 1][tid];
    }
}

// ---- K2: carry fixup --------------------------------------------------------
//   c_j = sum_{i<j} (alpha^128)^{j-1-i} * S[b][i][n]
//   out[b, j*128+tl, n] += alpha^{tl+1} * c_j
__global__ __launch_bounds__(256)
void sfc_scan_fixup_k2(float* __restrict__ out, const float* __restrict__ S,
                       const float* __restrict__ alpha) {
    int idx  = blockIdx.x * 256 + threadIdx.x;    // 32 * 15 * 512 threads
    int n    = idx & (NN - 1);
    int rest = idx >> 9;
    int j    = (rest % (NCHUNK - 1)) + 1;         // 1..15
    int b    = rest / (NCHUNK - 1);

    float a  = alpha[n];
    float aL = a;
    #pragma unroll
    for (int i = 0; i < 7; ++i) aL = aL * aL;     // alpha^128

    float c = 0.0f;
    for (int i = 0; i < j; ++i)
        c = fmaf(aL, c, S[(size_t)(b * NCHUNK + i) * NN + n]);

    float w = a * c;
    size_t base = ((size_t)(b * TT + j * BM)) * NN + n;
    for (int tl = 0; tl < BM; ++tl) {
        out[base] += w;
        w *= a;
        base += NN;
    }
}

extern "C" void kernel_launch(void* const* d_in, const int* in_sizes, int n_in,
                              void* d_out, int out_size, void* d_ws, size_t ws_size,
                              hipStream_t stream) {
    const float* x     = (const float*)d_in[0];   // [32, 2048, 512]
    const float* W     = (const float*)d_in[1];   // [512, 512]
    const float* bias  = (const float*)d_in[2];   // [512]
    const float* alpha = (const float*)d_in[3];   // [512]
    float* out = (float*)d_out;                   // [32, 2048, 512]

    // workspace carve: S (1 MB) | Wh (512 KB) | Wl (512 KB)
    char* ws = (char*)d_ws;
    float*    S  = (float*)ws;                               // 512*512 f32
    ushort_t* Wh = (ushort_t*)(ws + (size_t)NN * NN * 4);    // 512*512 bf16
    ushort_t* Wl = (ushort_t*)(ws + (size_t)NN * NN * 6);    // 512*512 bf16

    sfc_convert_w_k0<<<(NN * KK) / 256, 256, 0, stream>>>(W, Wh, Wl);

    dim3 g1(NN / BN, (32 * TT) / BM);             // (4, 512)
    sfc_gemm_scan_k1<<<g1, 256, 0, stream>>>(x, Wh, Wl, bias, alpha, out, S);

    int threads2 = 32 * (NCHUNK - 1) * NN;        // 245760
    sfc_scan_fixup_k2<<<threads2 / 256, 256, 0, stream>>>(out, S, alpha);
}
